// SBD_50740743635768
// MI455X (gfx1250) — matrix operand order repaired and checked by reading, not compiled, not run on hardware
//
#include <hip/hip_runtime.h>
#include <math.h>

#define BB   2
#define NN   16384
#define CC   256
#define MM   300
#define MP   320        // padded query/state rows per batch (multiple of 64)
#define HIDN 1024
#define NHEAD 8
#define NCLS 80
#define NCLSP 128       // padded cls columns (multiple of 64)
#define MAXD 100

typedef __attribute__((ext_vector_type(16))) __bf16 v16bf;
typedef __attribute__((ext_vector_type(8)))  __bf16 v8bf;
typedef __attribute__((ext_vector_type(8)))  float  v8f;

#define GF_RELU   1
#define GF_RESID  2
#define GF_OUTB16 4
#define GF_OUTVT  8

// ------ weight transpose + cast: W[K,N] f32 -> Wt[Npad,K] bf16, zero pad ----
__global__ void k_twt(const float* __restrict__ W, __bf16* __restrict__ Wt,
                      int K, int N, int Npad) {
  int i = blockIdx.x * blockDim.x + threadIdx.x;
  if (i >= Npad * K) return;
  int n = i / K, k = i % K;
  Wt[i] = (n < N) ? (__bf16)W[(size_t)k * N + n] : (__bf16)0.f;
}

// ---------------- DOD scores + feats f32->bf16 ------------------------------
__global__ void k_dod(const float* __restrict__ feats, const float* __restrict__ wdod,
                      const float* __restrict__ bdod, __bf16* __restrict__ featsb,
                      float* __restrict__ scores, int rows) {
  int wid = (blockIdx.x * blockDim.x + threadIdx.x) >> 5;
  int lane = threadIdx.x & 31;
  if (wid >= rows) return;
  const float* f = feats + (size_t)wid * CC;
  __bf16* fb = featsb + (size_t)wid * CC;
  float s = 0.f;
#pragma unroll
  for (int e = 0; e < 8; e++) {
    int c = lane * 8 + e;
    float v = f[c];
    fb[c] = (__bf16)v;
    s += v * wdod[c];
  }
#pragma unroll
  for (int off = 16; off >= 1; off >>= 1) s += __shfl_xor(s, off, 32);
  if (lane == 0) scores[wid] = s + bdod[0];
}

// ---------------- iterative top-k (mutates score buffer) --------------------
__global__ void k_topk(float* __restrict__ sc, int len, int k,
                       int* __restrict__ oidx, float* __restrict__ oval) {
  int b = blockIdx.x;
  float* s = sc + (size_t)b * len;
  __shared__ float bv[256];
  __shared__ int   bi[256];
  for (int it = 0; it < k; ++it) {
    float best = -3.0e38f; int bidx = 0x7fffffff;
    for (int i = threadIdx.x; i < len; i += 256) {
      float v = s[i];
      if (v > best || (v == best && i < bidx)) { best = v; bidx = i; }
    }
    bv[threadIdx.x] = best; bi[threadIdx.x] = bidx;
    __syncthreads();
    for (int off = 128; off; off >>= 1) {
      if ((int)threadIdx.x < off) {
        float v2 = bv[threadIdx.x + off]; int i2 = bi[threadIdx.x + off];
        if (v2 > bv[threadIdx.x] || (v2 == bv[threadIdx.x] && i2 < bi[threadIdx.x])) {
          bv[threadIdx.x] = v2; bi[threadIdx.x] = i2;
        }
      }
      __syncthreads();
    }
    if (threadIdx.x == 0) {
      oidx[b * k + it] = bi[0];
      oval[b * k + it] = bv[0];
      s[bi[0]] = -3.0e38f;
    }
    __syncthreads();
  }
}

// ---------------- gather selected feats/anchors (zero-pad rows 300..319) ----
__global__ void k_gather(const float* __restrict__ feats, const float* __restrict__ anchors,
                         const int* __restrict__ selidx, float* __restrict__ selfeat,
                         float* __restrict__ selanch) {
  int wid = (blockIdx.x * blockDim.x + threadIdx.x) >> 5;
  int lane = threadIdx.x & 31;
  if (wid >= BB * MP) return;
  int b = wid / MP, i = wid % MP;
  float* sf = selfeat + (size_t)wid * CC;
  if (i < MM) {
    int idx = selidx[b * MM + i];
    const float* src = feats + ((size_t)b * NN + idx) * CC;
#pragma unroll
    for (int e = 0; e < 8; e++) sf[lane * 8 + e] = src[lane * 8 + e];
    if (lane < 4) selanch[wid * 4 + lane] = anchors[((size_t)b * NN + idx) * 4 + lane];
  } else {
#pragma unroll
    for (int e = 0; e < 8; e++) sf[lane * 8 + e] = 0.f;
    if (lane < 4) selanch[wid * 4 + lane] = 0.f;
  }
}

// ---------------- AE layer1: relu(anchors[*,4] @ w1[4,256] + b1) ------------
__global__ void k_aeh(const float* __restrict__ selanch, const float* __restrict__ w1,
                      const float* __restrict__ b1, float* __restrict__ out) {
  int wid = (blockIdx.x * blockDim.x + threadIdx.x) >> 5;
  int lane = threadIdx.x & 31;
  if (wid >= BB * MP) return;
  float a0 = selanch[wid * 4 + 0], a1 = selanch[wid * 4 + 1];
  float a2 = selanch[wid * 4 + 2], a3 = selanch[wid * 4 + 3];
#pragma unroll
  for (int e = 0; e < 8; e++) {
    int n = lane * 8 + e;
    float v = b1[n] + a0 * w1[n] + a1 * w1[CC + n] + a2 * w1[2 * CC + n] + a3 * w1[3 * CC + n];
    out[(size_t)wid * CC + n] = fmaxf(v, 0.f);
  }
}

// ---------------- generic bf16 WMMA GEMM ------------------------------------
// C[M, Ncols] = A[M,K](f32->bf16) * Wt[Ncols,K](bf16) (+bias<Nbias)(relu)(+resid)
// Ncols = gridDim.y*64 (weights pre-padded, so NO bounds checks in hot loop).
// K must be a multiple of 64. flags: OUTB16 -> Cb[row*ldc+n];
// OUTVT -> Cb[((b*H+h)*32+d)*Lout + i], b=row/Lin, i=row%Lin, h=n>>5, d=n&31.
__global__ void k_gemm(const float* __restrict__ A, int lda,
                       const __bf16* __restrict__ Wt, int K,
                       const float* __restrict__ bias, int Nbias, int flags,
                       float* __restrict__ Cf, __bf16* __restrict__ Cb, int ldc,
                       const float* __restrict__ resid, int Lin, int Lout) {
  const int lane = threadIdx.x & 31;
  const int hf   = lane >> 4;
  const int sub  = lane & 15;
  const int m0 = blockIdx.x * 16;
  const int n0 = blockIdx.y * 64;
  v8f acc[4];
#pragma unroll
  for (int t = 0; t < 4; t++)
#pragma unroll
    for (int r = 0; r < 8; r++) acc[t][r] = 0.f;

  const float* ap = A + (size_t)(m0 + sub) * lda;
  for (int k0 = 0; k0 < K; k0 += 64) {
    v16bf af0, af1;
#pragma unroll
    for (int e = 0; e < 8; e++) {
      af0[e]     = (__bf16)ap[k0 + hf * 8 + e];
      af0[e + 8] = (__bf16)ap[k0 + 16 + hf * 8 + e];
      af1[e]     = (__bf16)ap[k0 + 32 + hf * 8 + e];
      af1[e + 8] = (__bf16)ap[k0 + 48 + hf * 8 + e];
    }
#pragma unroll
    for (int t = 0; t < 4; t++) {
      const __bf16* wp = Wt + (size_t)(n0 + t * 16 + sub) * K + k0 + hf * 8;
      v8bf w0 = *(const v8bf*)wp;
      v8bf w1 = *(const v8bf*)(wp + 16);
      v8bf w2 = *(const v8bf*)(wp + 32);
      v8bf w3 = *(const v8bf*)(wp + 48);
      v16bf b0, b1;
#pragma unroll
      for (int e = 0; e < 8; e++) { b0[e] = w0[e]; b0[e + 8] = w1[e]; b1[e] = w2[e]; b1[e + 8] = w3[e]; }
      acc[t] = __builtin_amdgcn_wmma_f32_16x16x32_bf16(false, af0, false, b0,
                                                       (short)0, acc[t], false, false);
      acc[t] = __builtin_amdgcn_wmma_f32_16x16x32_bf16(false, af1, false, b1,
                                                       (short)0, acc[t], false, false);
    }
  }
#pragma unroll
  for (int t = 0; t < 4; t++) {
    int n = n0 + t * 16 + sub;
    float bvl = 0.f;
    if (n0 + t * 16 < Nbias) bvl = bias[n];   // uniform scalar guard per tile
#pragma unroll
    for (int r = 0; r < 8; r++) {
      int row = m0 + hf * 8 + r;
      float v = acc[t][r] + bvl;
      if (flags & GF_RELU)  v = fmaxf(v, 0.f);
      if (flags & GF_RESID) v += resid[(size_t)row * ldc + n];
      if (flags & GF_OUTVT) {
        int b = row / Lin, i = row % Lin;
        int h = n >> 5, d = n & 31;
        Cb[((size_t)((b * NHEAD + h) * 32 + d)) * Lout + i] = (__bf16)v;
      } else if (flags & GF_OUTB16) {
        Cb[(size_t)row * ldc + n] = (__bf16)v;
      } else {
        Cf[(size_t)row * ldc + n] = v;
      }
    }
  }
}

// ---------------- flash attention (one wave = 16 queries x 1 head) ----------
// Q:[B,LQ,256]bf16  K:[B,LK,256]bf16  Vt:[B,H,32,LK]bf16  O:[B,LQ,256]f32
// LK must be a multiple of 64; kv >= kvlen masked to -inf before softmax.
__global__ void k_flash(const __bf16* __restrict__ Q, const __bf16* __restrict__ Kb,
                        const __bf16* __restrict__ Vt, float* __restrict__ O,
                        int LQ, int LK, int kvlen, float scale) {
  __shared__ __bf16 pshm[16 * 64];
  const int lane = threadIdx.x & 31;
  const int hf   = lane >> 4;
  const int sub  = lane & 15;
  const int qtiles = LQ >> 4;
  const int qt = blockIdx.x % qtiles;
  const int h  = (blockIdx.x / qtiles) & (NHEAD - 1);
  const int b  = blockIdx.x / (qtiles * NHEAD);

  const __bf16* qp = Q + ((size_t)(b * LQ + qt * 16 + sub)) * CC + h * 32;
  v16bf aq;
#pragma unroll
  for (int e = 0; e < 8; e++) { aq[e] = qp[hf * 16 + e]; aq[e + 8] = qp[hf * 16 + 8 + e]; }

  v8f acc0, acc1, zc;
  float mrun[8], lrun[8];
#pragma unroll
  for (int r = 0; r < 8; r++) { acc0[r] = 0.f; acc1[r] = 0.f; zc[r] = 0.f; mrun[r] = -1e30f; lrun[r] = 0.f; }

  const __bf16* kbase = Kb + (size_t)b * LK * CC + h * 32 + hf * 16;
  const __bf16* vbase = Vt + ((size_t)(b * NHEAD + h) * 32) * LK;

  for (int n0 = 0; n0 < LK; n0 += 64) {
    if (n0 + 64 < LK) {
      __builtin_prefetch(kbase + (size_t)(n0 + 64 + sub) * CC, 0, 1);
      __builtin_prefetch(kbase + (size_t)(n0 + 96 + sub) * CC, 0, 1);
    }
    // ---- S = Q K^T for 64 keys (4 WMMA) ----
    v8f s0, s1, s2, s3;
    {
      v16bf bk;
      const __bf16* kp;
#define LOADK(J, DST)                                                            \
      kp = kbase + (size_t)(n0 + (J)*16 + sub) * CC;                             \
      { v8bf w0 = *(const v8bf*)kp; v8bf w1 = *(const v8bf*)(kp + 8);            \
        _Pragma("unroll")                                                        \
        for (int e = 0; e < 8; e++) { bk[e] = w0[e]; bk[e + 8] = w1[e]; } }      \
      DST = __builtin_amdgcn_wmma_f32_16x16x32_bf16(false, aq, false, bk,        \
                                                    (short)0, zc, false, false);
      LOADK(0, s0) LOADK(1, s1) LOADK(2, s2) LOADK(3, s3)
#undef LOADK
    }
    const bool ok0 = (n0 + sub) < kvlen;
    const bool ok1 = (n0 + 16 + sub) < kvlen;
    const bool ok2 = (n0 + 32 + sub) < kvlen;
    const bool ok3 = (n0 + 48 + sub) < kvlen;
    // ---- online softmax for 8 rows ----
#pragma unroll
    for (int r = 0; r < 8; r++) {
      float f0 = ok0 ? s0[r] * scale : -1e30f;
      float f1 = ok1 ? s1[r] * scale : -1e30f;
      float f2 = ok2 ? s2[r] * scale : -1e30f;
      float f3 = ok3 ? s3[r] * scale : -1e30f;
      float cm = fmaxf(fmaxf(f0, f1), fmaxf(f2, f3));
#pragma unroll
      for (int off = 8; off >= 1; off >>= 1) cm = fmaxf(cm, __shfl_xor(cm, off, 32));
      float mnew = fmaxf(mrun[r], cm);
      float sc_ = __expf(mrun[r] - mnew);
      mrun[r] = mnew;
      acc0[r] *= sc_; acc1[r] *= sc_; lrun[r] *= sc_;
      float p0 = __expf(f0 - mnew), p1 = __expf(f1 - mnew);
      float p2 = __expf(f2 - mnew), p3 = __expf(f3 - mnew);
      float rs = (p0 + p1) + (p2 + p3);
#pragma unroll
      for (int off = 8; off >= 1; off >>= 1) rs += __shfl_xor(rs, off, 32);
      lrun[r] += rs;
      __bf16* pr = pshm + (r + hf * 8) * 64 + sub;
      pr[0]  = (__bf16)p0;
      pr[16] = (__bf16)p1;
      pr[32] = (__bf16)p2;
      pr[48] = (__bf16)p3;
    }
    __syncthreads();
    // ---- reload P as two A-fragments (k = kv 0..31 and 32..63) ----
    v16bf ap0, ap1;
    {
      const __bf16* pp = pshm + sub * 64;
#pragma unroll
      for (int e = 0; e < 8; e++) {
        ap0[e]     = pp[hf * 16 + e];
        ap0[e + 8] = pp[hf * 16 + 8 + e];
        ap1[e]     = pp[32 + hf * 16 + e];
        ap1[e + 8] = pp[32 + hf * 16 + 8 + e];
      }
    }
    __syncthreads();
    // ---- O += P V for both 16-wide d halves (4 WMMA) ----
#pragma unroll
    for (int g = 0; g < 2; g++) {
      const __bf16* vp0 = vbase + (size_t)sub * LK + n0 + g * 32 + hf * 16;
      const __bf16* vp1 = vbase + (size_t)(16 + sub) * LK + n0 + g * 32 + hf * 16;
      v8bf a0 = *(const v8bf*)vp0, a1 = *(const v8bf*)(vp0 + 8);
      v8bf c0 = *(const v8bf*)vp1, c1 = *(const v8bf*)(vp1 + 8);
      v16bf bv0, bv1;
#pragma unroll
      for (int e = 0; e < 8; e++) { bv0[e] = a0[e]; bv0[e + 8] = a1[e]; bv1[e] = c0[e]; bv1[e + 8] = c1[e]; }
      if (g == 0) {
        acc0 = __builtin_amdgcn_wmma_f32_16x16x32_bf16(false, ap0, false, bv0, (short)0, acc0, false, false);
        acc1 = __builtin_amdgcn_wmma_f32_16x16x32_bf16(false, ap0, false, bv1, (short)0, acc1, false, false);
      } else {
        acc0 = __builtin_amdgcn_wmma_f32_16x16x32_bf16(false, ap1, false, bv0, (short)0, acc0, false, false);
        acc1 = __builtin_amdgcn_wmma_f32_16x16x32_bf16(false, ap1, false, bv1, (short)0, acc1, false, false);
      }
    }
  }
  float* op = O + ((size_t)(b * LQ + qt * 16)) * CC + h * 32;
#pragma unroll
  for (int r = 0; r < 8; r++) {
    int row = r + hf * 8;
    float inv = 1.f / lrun[r];
    op[(size_t)row * CC + sub]      = acc0[r] * inv;
    op[(size_t)row * CC + 16 + sub] = acc1[r] * inv;
  }
}

// ---------------- layernorm over 256, in-place (one wave per row) -----------
__global__ void k_ln(float* __restrict__ x, const float* __restrict__ g,
                     const float* __restrict__ bb, int rows) {
  int wid = (blockIdx.x * blockDim.x + threadIdx.x) >> 5;
  int lane = threadIdx.x & 31;
  if (wid >= rows) return;
  float* r = x + (size_t)wid * CC;
  float v[8], s = 0.f, s2 = 0.f;
#pragma unroll
  for (int e = 0; e < 8; e++) { v[e] = r[lane * 8 + e]; s += v[e]; s2 += v[e] * v[e]; }
#pragma unroll
  for (int off = 16; off >= 1; off >>= 1) { s += __shfl_xor(s, off, 32); s2 += __shfl_xor(s2, off, 32); }
  float mu = s * (1.f / CC);
  float var = s2 * (1.f / CC) - mu * mu;
  float inv = rsqrtf(var + 1e-5f);
#pragma unroll
  for (int e = 0; e < 8; e++) {
    int c = lane * 8 + e;
    r[c] = (v[e] - mu) * inv * g[c] + bb[c];
  }
}

// ---------------- small head: hid[*,256] @ W[256,4] + b ---------------------
__global__ void k_boxw2(const float* __restrict__ hid, const float* __restrict__ w,
                        const float* __restrict__ bb, float* __restrict__ out, int rows) {
  int wid = (blockIdx.x * blockDim.x + threadIdx.x) >> 5;
  int lane = threadIdx.x & 31;
  if (wid >= rows) return;
  float a0 = 0, a1 = 0, a2 = 0, a3 = 0;
#pragma unroll
  for (int e = 0; e < 8; e++) {
    int k = lane * 8 + e;
    float hv = hid[(size_t)wid * CC + k];
    a0 += hv * w[k * 4 + 0]; a1 += hv * w[k * 4 + 1];
    a2 += hv * w[k * 4 + 2]; a3 += hv * w[k * 4 + 3];
  }
#pragma unroll
  for (int off = 16; off >= 1; off >>= 1) {
    a0 += __shfl_xor(a0, off, 32); a1 += __shfl_xor(a1, off, 32);
    a2 += __shfl_xor(a2, off, 32); a3 += __shfl_xor(a3, off, 32);
  }
  if (lane == 0) {
    out[wid * 4 + 0] = a0 + bb[0]; out[wid * 4 + 1] = a1 + bb[1];
    out[wid * 4 + 2] = a2 + bb[2]; out[wid * 4 + 3] = a3 + bb[3];
  }
}

// ---------------- box decode -------------------------------------------------
__global__ void k_boxdec(const float* __restrict__ deltas, const float* __restrict__ selanch,
                         float* __restrict__ boxes, int rows) {
  int r = blockIdx.x * blockDim.x + threadIdx.x;
  if (r >= rows) return;
  float d0 = deltas[r * 4 + 0], d1 = deltas[r * 4 + 1], d2 = deltas[r * 4 + 2], d3 = deltas[r * 4 + 3];
  float a0 = selanch[r * 4 + 0], a1 = selanch[r * 4 + 1], a2 = selanch[r * 4 + 2], a3 = selanch[r * 4 + 3];
  float cx = a0 + 0.1f * tanhf(d0);
  float cy = a1 + 0.1f * tanhf(d1);
  float w  = (0.02f + 0.3f * a2) * __expf(0.2f * tanhf(d2));
  float h  = (0.02f + 0.3f * a3) * __expf(0.2f * tanhf(d3));
  boxes[r * 4 + 0] = cx - 0.5f * w;
  boxes[r * 4 + 1] = cy - 0.5f * h;
  boxes[r * 4 + 2] = cx + 0.5f * w;
  boxes[r * 4 + 3] = cy + 0.5f * h;
}

// ---------------- sigmoid probs [B, 300*80] ---------------------------------
__global__ void k_probs(const float* __restrict__ cls, float* __restrict__ probs) {
  int i = blockIdx.x * blockDim.x + threadIdx.x;
  if (i >= BB * MM * NCLS) return;
  int b = i / (MM * NCLS);
  int r = i % (MM * NCLS);
  int m = r / NCLS, c = r % NCLS;
  float lg = cls[((size_t)(b * MP + m)) * NCLSP + c];
  probs[i] = 1.f / (1.f + __expf(-lg));
}

// ---------------- gather candidate boxes ------------------------------------
__global__ void k_gcand(const int* __restrict__ cidx, const float* __restrict__ boxes,
                        float* __restrict__ cboxes) {
  int i = blockIdx.x * blockDim.x + threadIdx.x;
  if (i >= BB * MM) return;
  int b = i / MM;
  int obj = cidx[i] / NCLS;
#pragma unroll
  for (int t = 0; t < 4; t++) cboxes[i * 4 + t] = boxes[((size_t)(b * MP + obj)) * 4 + t];
}

// ---------------- greedy NMS + mask ------------------------------------------
__global__ void k_nms(const float* __restrict__ cboxes, const float* __restrict__ cval,
                      float* __restrict__ masked) {
  int b = blockIdx.x;
  __shared__ float bx[MM][4];
  __shared__ float ar[MM];
  __shared__ int sup[MM];
  for (int i = threadIdx.x; i < MM; i += blockDim.x) {
#pragma unroll
    for (int j = 0; j < 4; j++) bx[i][j] = cboxes[((size_t)b * MM + i) * 4 + j];
    float w = fmaxf(bx[i][2] - bx[i][0], 0.f), h = fmaxf(bx[i][3] - bx[i][1], 0.f);
    ar[i] = w * h; sup[i] = 0;
  }
  __syncthreads();
  for (int i = 0; i < MM; i++) {
    if (!sup[i]) {
      for (int j = i + 1 + threadIdx.x; j < MM; j += blockDim.x) {
        float ltx = fmaxf(bx[i][0], bx[j][0]), lty = fmaxf(bx[i][1], bx[j][1]);
        float rbx = fminf(bx[i][2], bx[j][2]), rby = fminf(bx[i][3], bx[j][3]);
        float w = fmaxf(rbx - ltx, 0.f), h = fmaxf(rby - lty, 0.f);
        float inter = w * h;
        float iou = inter / (ar[i] + ar[j] - inter + 1e-6f);
        if (iou >= 0.5f) sup[j] = 1;
      }
    }
    __syncthreads();
  }
  for (int j = threadIdx.x; j < MM; j += blockDim.x)
    masked[b * MM + j] = sup[j] ? -1.0f : cval[b * MM + j];
}

// ---------------- final output assembly --------------------------------------
__global__ void k_final(const float* __restrict__ dval, const int* __restrict__ didx,
                        const float* __restrict__ cboxes, float* __restrict__ out) {
  int i = blockIdx.x * blockDim.x + threadIdx.x;
  if (i >= BB * MAXD) return;
  int b = i / MAXD;
  out[i * 5] = dval[i];
  int di = didx[i];
#pragma unroll
  for (int t = 0; t < 4; t++) out[i * 5 + 1 + t] = cboxes[((size_t)b * MM + di) * 4 + t];
}

// ============================================================================
extern "C" void kernel_launch(void* const* d_in, const int* in_sizes, int n_in,
                              void* d_out, int out_size, void* d_ws, size_t ws_size,
                              hipStream_t stream) {
  (void)in_sizes; (void)n_in; (void)out_size; (void)ws_size;
  const float* feats   = (const float*)d_in[0];
  const float* anchors = (const float*)d_in[1];
  const float* w_dod   = (const float*)d_in[2];
  const float* b_dod   = (const float*)d_in[3];
  const float* osi_w1  = (const float*)d_in[4];
  const float* osi_b1  = (const float*)d_in[5];
  const float* osi_w2  = (const float*)d_in[6];
  const float* osi_b2  = (const float*)d_in[7];
  const float* ae_w1   = (const float*)d_in[8];
  const float* ae_b1   = (const float*)d_in[9];
  const float* ae_w2   = (const float*)d_in[10];
  const float* ae_b2   = (const float*)d_in[11];
  const float* ca_wq = (const float*)d_in[12]; const float* ca_bq = (const float*)d_in[13];
  const float* ca_wk = (const float*)d_in[14]; const float* ca_bk = (const float*)d_in[15];
  const float* ca_wv = (const float*)d_in[16]; const float* ca_bv = (const float*)d_in[17];
  const float* ca_wo = (const float*)d_in[18]; const float* ca_bo = (const float*)d_in[19];
  const float* sa_wq = (const float*)d_in[20]; const float* sa_bq = (const float*)d_in[21];
  const float* sa_wk = (const float*)d_in[22]; const float* sa_bk = (const float*)d_in[23];
  const float* sa_wv = (const float*)d_in[24]; const float* sa_bv = (const float*)d_in[25];
  const float* sa_wo = (const float*)d_in[26]; const float* sa_bo = (const float*)d_in[27];
  const float* ffn_w1 = (const float*)d_in[28]; const float* ffn_b1 = (const float*)d_in[29];
  const float* ffn_w2 = (const float*)d_in[30]; const float* ffn_b2 = (const float*)d_in[31];
  const float* ln1_g = (const float*)d_in[32]; const float* ln1_b = (const float*)d_in[33];
  const float* ln2_g = (const float*)d_in[34]; const float* ln2_b = (const float*)d_in[35];
  const float* ln3_g = (const float*)d_in[36]; const float* ln3_b = (const float*)d_in[37];
  const float* cls_w1 = (const float*)d_in[38]; const float* cls_b1 = (const float*)d_in[39];
  const float* cls_w2 = (const float*)d_in[40]; const float* cls_b2 = (const float*)d_in[41];
  const float* box_w1 = (const float*)d_in[42]; const float* box_b1 = (const float*)d_in[43];
  const float* box_w2 = (const float*)d_in[44]; const float* box_b2 = (const float*)d_in[45];
  float* out = (float*)d_out;

  // ----- workspace bump allocator -----
  char* ws = (char*)d_ws;
  size_t off = 0;
  auto alloc = [&](size_t bytes) -> char* {
    char* p = ws + off;
    off += (bytes + 255) & ~(size_t)255;
    return p;
  };
  __bf16* featsb = (__bf16*)alloc((size_t)BB * NN * CC * 2);
  __bf16* kc     = (__bf16*)alloc((size_t)BB * NN * CC * 2);
  __bf16* vtc    = (__bf16*)alloc((size_t)BB * NN * CC * 2);
  float*  scores = (float*)alloc((size_t)BB * NN * 4);
  int*    selidx = (int*)alloc((size_t)BB * MM * 4);
  float*  selval = (float*)alloc((size_t)BB * MM * 4);
  float*  selfeat = (float*)alloc((size_t)BB * MP * CC * 4);
  float*  selanch = (float*)alloc((size_t)BB * MP * 4 * 4);
  float*  aeh     = (float*)alloc((size_t)BB * MP * CC * 4);
  float*  hid     = (float*)alloc((size_t)BB * MP * HIDN * 4);
  float*  states  = (float*)alloc((size_t)BB * MP * CC * 4);
  __bf16* qb      = (__bf16*)alloc((size_t)BB * MP * CC * 2);
  float*  attn    = (float*)alloc((size_t)BB * MP * CC * 4);
  __bf16* ks      = (__bf16*)alloc((size_t)BB * MP * CC * 2);
  __bf16* vts     = (__bf16*)alloc((size_t)BB * MP * CC * 2);
  float*  clsl    = (float*)alloc((size_t)BB * MP * NCLSP * 4);
  float*  deltas  = (float*)alloc((size_t)BB * MP * 4 * 4);
  float*  boxes   = (float*)alloc((size_t)BB * MP * 4 * 4);
  float*  probs   = (float*)alloc((size_t)BB * MM * NCLS * 4);
  int*    cidx    = (int*)alloc((size_t)BB * MM * 4);
  float*  cval    = (float*)alloc((size_t)BB * MM * 4);
  float*  cboxes  = (float*)alloc((size_t)BB * MM * 4 * 4);
  float*  masked  = (float*)alloc((size_t)BB * MM * 4);
  int*    didx    = (int*)alloc((size_t)BB * MAXD * 4);
  float*  dval    = (float*)alloc((size_t)BB * MAXD * 4);
  // transposed bf16 weights (padded to multiple-of-64 output columns)
  __bf16* t_osi1 = (__bf16*)alloc((size_t)CC * HIDN * 2);
  __bf16* t_osi2 = (__bf16*)alloc((size_t)HIDN * CC * 2);
  __bf16* t_ae2  = (__bf16*)alloc((size_t)CC * CC * 2);
  __bf16* t_cawq = (__bf16*)alloc((size_t)CC * CC * 2);
  __bf16* t_cawk = (__bf16*)alloc((size_t)CC * CC * 2);
  __bf16* t_cawv = (__bf16*)alloc((size_t)CC * CC * 2);
  __bf16* t_cawo = (__bf16*)alloc((size_t)CC * CC * 2);
  __bf16* t_sawq = (__bf16*)alloc((size_t)CC * CC * 2);
  __bf16* t_sawk = (__bf16*)alloc((size_t)CC * CC * 2);
  __bf16* t_sawv = (__bf16*)alloc((size_t)CC * CC * 2);
  __bf16* t_sawo = (__bf16*)alloc((size_t)CC * CC * 2);
  __bf16* t_ffn1 = (__bf16*)alloc((size_t)CC * HIDN * 2);
  __bf16* t_ffn2 = (__bf16*)alloc((size_t)HIDN * CC * 2);
  __bf16* t_cls1 = (__bf16*)alloc((size_t)CC * CC * 2);
  __bf16* t_cls2 = (__bf16*)alloc((size_t)NCLSP * CC * 2);   // padded 128 x 256
  __bf16* t_box1 = (__bf16*)alloc((size_t)CC * CC * 2);

  auto twt = [&](const float* W, __bf16* Wt, int K, int N, int Npad) {
    int tot = Npad * K;
    k_twt<<<(tot + 255) / 256, 256, 0, stream>>>(W, Wt, K, N, Npad);
  };
  // Ncols: padded output columns (multiple of 64); Nbias: true column count.
  auto gemm = [&](const float* A, int lda, const __bf16* Wt, int K, const float* bias,
                  int Nbias, int Ncols, int Mrows, int flags, float* Cf, __bf16* Cb,
                  int ldc, const float* resid, int Lin, int Lout) {
    dim3 grid(Mrows / 16, Ncols / 64);
    k_gemm<<<grid, 32, 0, stream>>>(A, lda, Wt, K, bias, Nbias, flags, Cf, Cb, ldc,
                                    resid, Lin, Lout);
  };

  const float att_scale = 0.1767766952966369f;  // 1/sqrt(32)
  const int ROWS = BB * MP;                     // 640 padded state rows

  // ----- weight prep -----
  twt(osi_w1, t_osi1, CC, HIDN, HIDN);
  twt(osi_w2, t_osi2, HIDN, CC, CC);
  twt(ae_w2, t_ae2, CC, CC, CC);
  twt(ca_wq, t_cawq, CC, CC, CC); twt(ca_wk, t_cawk, CC, CC, CC);
  twt(ca_wv, t_cawv, CC, CC, CC); twt(ca_wo, t_cawo, CC, CC, CC);
  twt(sa_wq, t_sawq, CC, CC, CC); twt(sa_wk, t_sawk, CC, CC, CC);
  twt(sa_wv, t_sawv, CC, CC, CC); twt(sa_wo, t_sawo, CC, CC, CC);
  twt(ffn_w1, t_ffn1, CC, HIDN, HIDN);
  twt(ffn_w2, t_ffn2, HIDN, CC, CC);
  twt(cls_w1, t_cls1, CC, CC, CC);
  twt(cls_w2, t_cls2, CC, NCLS, NCLSP);
  twt(box_w1, t_box1, CC, CC, CC);

  // ----- DOD scores + bf16 cast of feats -----
  k_dod<<<(BB * NN) / 8, 256, 0, stream>>>(feats, w_dod, b_dod, featsb, scores, BB * NN);
  k_topk<<<BB, 256, 0, stream>>>(scores, NN, MM, selidx, selval);
  k_gather<<<(ROWS + 7) / 8, 256, 0, stream>>>(feats, anchors, selidx, selfeat, selanch);

  // ----- initial states: OSI(selfeat) + AE(selanch) -----
  k_aeh<<<(ROWS + 7) / 8, 256, 0, stream>>>(selanch, ae_w1, ae_b1, aeh);
  gemm(selfeat, CC, t_osi1, CC, osi_b1, HIDN, HIDN, ROWS, GF_RELU, hid, nullptr, HIDN, nullptr, 0, 0);
  gemm(hid, HIDN, t_osi2, HIDN, osi_b2, CC, CC, ROWS, 0, states, nullptr, CC, nullptr, 0, 0);
  gemm(aeh, CC, t_ae2, CC, ae_b2, CC, CC, ROWS, GF_RESID, states, nullptr, CC, states, 0, 0);

  // ----- cross-attention K/V projections (loop-invariant: compute ONCE) -----
  gemm(feats, CC, t_cawk, CC, ca_bk, CC, CC, BB * NN, GF_OUTB16, nullptr, kc, CC, nullptr, 0, 0);
  gemm(feats, CC, t_cawv, CC, ca_bv, CC, CC, BB * NN, GF_OUTVT, nullptr, vtc, CC, nullptr, NN, NN);

  // ----- iterative update layers -----
  for (int it = 0; it < 6; ++it) {
    // cross-attn
    gemm(states, CC, t_cawq, CC, ca_bq, CC, CC, ROWS, GF_OUTB16, nullptr, qb, CC, nullptr, 0, 0);
    k_flash<<<BB * NHEAD * (MP / 16), 32, 0, stream>>>(qb, kc, vtc, attn, MP, NN, NN, att_scale);
    gemm(attn, CC, t_cawo, CC, ca_bo, CC, CC, ROWS, GF_RESID, states, nullptr, CC, states, 0, 0);
    k_ln<<<(ROWS + 7) / 8, 256, 0, stream>>>(states, ln1_g, ln1_b, ROWS);
    // self-attn
    gemm(states, CC, t_sawk, CC, sa_bk, CC, CC, ROWS, GF_OUTB16, nullptr, ks, CC, nullptr, 0, 0);
    gemm(states, CC, t_sawv, CC, sa_bv, CC, CC, ROWS, GF_OUTVT, nullptr, vts, CC, nullptr, MP, MP);
    gemm(states, CC, t_sawq, CC, sa_bq, CC, CC, ROWS, GF_OUTB16, nullptr, qb, CC, nullptr, 0, 0);
    k_flash<<<BB * NHEAD * (MP / 16), 32, 0, stream>>>(qb, ks, vts, attn, MP, MP, MM, att_scale);
    gemm(attn, CC, t_sawo, CC, sa_bo, CC, CC, ROWS, GF_RESID, states, nullptr, CC, states, 0, 0);
    k_ln<<<(ROWS + 7) / 8, 256, 0, stream>>>(states, ln2_g, ln2_b, ROWS);
    // FFN
    gemm(states, CC, t_ffn1, CC, ffn_b1, HIDN, HIDN, ROWS, GF_RELU, hid, nullptr, HIDN, nullptr, 0, 0);
    gemm(hid, HIDN, t_ffn2, HIDN, ffn_b2, CC, CC, ROWS, GF_RESID, states, nullptr, CC, states, 0, 0);
    k_ln<<<(ROWS + 7) / 8, 256, 0, stream>>>(states, ln3_g, ln3_b, ROWS);
  }

  // ----- CLS head -----
  gemm(states, CC, t_cls1, CC, cls_b1, CC, CC, ROWS, GF_RELU, hid, nullptr, CC, nullptr, 0, 0);
  gemm(hid, CC, t_cls2, CC, cls_b2, NCLS, NCLSP, ROWS, 0, clsl, nullptr, NCLSP, nullptr, 0, 0);
  // ----- BOX head -----
  gemm(states, CC, t_box1, CC, box_b1, CC, CC, ROWS, GF_RELU, hid, nullptr, CC, nullptr, 0, 0);
  k_boxw2<<<(ROWS + 7) / 8, 256, 0, stream>>>(hid, box_w2, box_b2, deltas, ROWS);
  k_boxdec<<<(ROWS + 255) / 256, 256, 0, stream>>>(deltas, selanch, boxes, ROWS);

  // ----- candidates, NMS, final top-100 -----
  k_probs<<<(BB * MM * NCLS + 255) / 256, 256, 0, stream>>>(clsl, probs);
  k_topk<<<BB, 256, 0, stream>>>(probs, MM * NCLS, MM, cidx, cval);
  k_gcand<<<(BB * MM + 255) / 256, 256, 0, stream>>>(cidx, boxes, cboxes);
  k_nms<<<BB, 256, 0, stream>>>(cboxes, cval, masked);
  k_topk<<<BB, 256, 0, stream>>>(masked, MM, MAXD, didx, dval);
  k_final<<<(BB * MAXD + 255) / 256, 256, 0, stream>>>(dval, didx, cboxes, out);
}
